// MemoryGraph_64226940944454
// MI455X (gfx1250) — matrix operand order repaired and checked by reading, not compiled, not run on hardware
//
#include <hip/hip_runtime.h>
#include <hip/hip_bf16.h>
#include <cstdint>

// Problem constants (fixed by reference setup_inputs)
#define BS   8
#define TT   32
#define CCH  64
#define NN   4096
#define KKN  32
#define DD   64
#define HHD  64
#define NBB  4
#define BSZB 8
#define BPGG 4
#define FIVED 320

typedef __attribute__((ext_vector_type(16))) __bf16       v16bf;
typedef __attribute__((ext_vector_type(8)))  float        v8f;
typedef __attribute__((ext_vector_type(8)))  unsigned int v8u;

struct U4x2 { uint4 lo, hi; };

__device__ __forceinline__ unsigned int pack_bf16(float a, float b) {
    unsigned ua = __builtin_bit_cast(unsigned, a);
    unsigned ub = __builtin_bit_cast(unsigned, b);
    ua += 0x7FFFu + ((ua >> 16) & 1u);   // round-to-nearest-even
    ub += 0x7FFFu + ((ub >> 16) & 1u);
    return (ua >> 16) | (ub & 0xFFFF0000u);
}

__device__ __forceinline__ float sigmoidf_(float x) { return 1.0f / (1.0f + expf(-x)); }

// ---------------------------------------------------------------------------
// Phase A: per-neuron modulation MLP via v_wmma_f32_16x16x32_bf16.
// One wave per neuron n. A = mod_input (8x320, rows 8..15 of the tile are
// duplicates -> only C rows 0..7 read). B = fc1_w[n] tiles (32x16).
// ---------------------------------------------------------------------------
#define AW 4   // waves per block

__global__ __launch_bounds__(AW*32) void phaseA_kernel(
    const float* __restrict__ h_in,  const float* __restrict__ tprim,
    const float* __restrict__ tkey,  const float* __restrict__ prim,
    const float* __restrict__ keyp,  const float* __restrict__ dlog,
    const float* __restrict__ fc1w,  const float* __restrict__ fc1b,
    const float* __restrict__ fc2w,  const float* __restrict__ fc2b,
    const float* __restrict__ mll,
    float* __restrict__ eff_prim, float* __restrict__ eff_key,
    float* __restrict__ eff_dec)
{
    __shared__ __align__(16) unsigned short lds_a[AW][BS*FIVED]; // bf16 mod_input
    __shared__ float lds_x[AW][BS*HHD];
    __shared__ float lds_o[AW][BS*3];

    const int wave = threadIdx.x >> 5;
    const int lane = threadIdx.x & 31;
    const int n = blockIdx.x * AW + wave;

    // ---- stage mod_input = [h, trace_prim, trace_key, primitives, key_p] ----
    for (int b = 0; b < BS; ++b) {
        const size_t bn = ((size_t)b*NN + n) * DD;
        const float* segs[5] = { h_in + bn, tprim + bn, tkey + bn,
                                 prim + (size_t)n*DD, keyp + (size_t)n*DD };
        #pragma unroll
        for (int s = 0; s < 5; ++s) {
            float2 v = reinterpret_cast<const float2*>(segs[s])[lane];
            *reinterpret_cast<unsigned int*>(&lds_a[wave][b*FIVED + s*DD + lane*2])
                = pack_bf16(v.x, v.y);
        }
    }
    __syncthreads();

    // ---- GEMM: (16x320)x(320x64) in 10 k-steps x 4 h-tiles ----
    v8f acc[4];
    #pragma unroll
    for (int i = 0; i < 4; ++i)
        #pragma unroll
        for (int r = 0; r < 8; ++r) acc[i][r] = 0.0f;

    const int kbaseA = (lane < 16) ? 0 : 8;   // ISA 16-bit A layout
    const int arow   = lane & 7;              // rows 8..15 mirror 0..7 (unused)
    const int koffB  = (lane < 16) ? 0 : 8;   // ISA 16-bit B layout (matches A)
    const int hlane  = lane & 15;

    for (int kk = 0; kk < 10; ++kk) {
        const unsigned short* ap = &lds_a[wave][arow*FIVED + kk*32 + kbaseA];
        U4x2 au;
        au.lo = *reinterpret_cast<const uint4*>(ap);        // K = kbase..kbase+7
        au.hi = *reinterpret_cast<const uint4*>(ap + 16);   // K = kbase+16..+23
        v16bf A = __builtin_bit_cast(v16bf, au);

        const float* wbase = fc1w + ((size_t)n*FIVED + kk*32 + koffB) * HHD;
        #pragma unroll
        for (int ht = 0; ht < 4; ++ht) {
            const float* wp = wbase + ht*16 + hlane;
            v8u pk;
            #pragma unroll
            for (int j = 0; j < 8; ++j) {
                float f0 = wp[(size_t)(2*j   + (j >= 4 ? 8 : 0)) * HHD];
                float f1 = wp[(size_t)(2*j+1 + (j >= 4 ? 8 : 0)) * HHD];
                pk[j] = pack_bf16(f0, f1);
            }
            v16bf B = __builtin_bit_cast(v16bf, pk);
            acc[ht] = __builtin_amdgcn_wmma_f32_16x16x32_bf16(
                false, A, false, B, (short)0, acc[ht], false, false);
        }
    }

    // ---- x = tanh(gemm + fc1_b); lanes 0..15 hold b=r at VGPR r ----
    if (lane < 16) {
        #pragma unroll
        for (int ht = 0; ht < 4; ++ht) {
            int hcol = ht*16 + lane;
            float bias = fc1b[(size_t)n*HHD + hcol];
            #pragma unroll
            for (int r = 0; r < 8; ++r)
                lds_x[wave][r*HHD + hcol] = tanhf(acc[ht][r] + bias);
        }
    }
    __syncthreads();

    // ---- fc2: 24 (b,o) dot products of length 64 ----
    if (lane < 24) {
        int b = lane / 3, o = lane % 3;
        float a = fc2b[(size_t)n*3 + o];
        const float* w2 = fc2w + (size_t)n*HHD*3 + o;
        const float* xr = &lds_x[wave][b*HHD];
        #pragma unroll 8
        for (int hh = 0; hh < HHD; ++hh) a += xr[hh] * w2[(size_t)hh*3];
        lds_o[wave][b*3 + o] = a;
    }
    __syncthreads();

    // ---- gates -> eff_prim / eff_key / eff_decay ----
    float mod_lr = sigmoidf_(mll[0]);
    float dl = dlog[n];
    float2 p2 = reinterpret_cast<const float2*>(prim + (size_t)n*DD)[lane];
    float2 k2 = reinterpret_cast<const float2*>(keyp + (size_t)n*DD)[lane];
    for (int b = 0; b < BS; ++b) {
        const size_t base = ((size_t)b*NN + n) * DD;
        float2 tp = reinterpret_cast<const float2*>(tprim + base)[lane];
        float2 tk = reinterpret_cast<const float2*>(tkey  + base)[lane];
        float np = tp.x*tp.x + tp.y*tp.y;
        float nk = tk.x*tk.x + tk.y*tk.y;
        #pragma unroll
        for (int off = 16; off >= 1; off >>= 1) {
            np += __shfl_xor(np, off, 32);
            nk += __shfl_xor(nk, off, 32);
        }
        float ip = 1.0f / fmaxf(sqrtf(np), 1e-8f);
        float ik = 1.0f / fmaxf(sqrtf(nk), 1e-8f);
        float gp = tanhf(lds_o[wave][b*3 + 0]);
        float gk = tanhf(lds_o[wave][b*3 + 1]);
        float dm = lds_o[wave][b*3 + 2];
        float sp = mod_lr * gp * ip;
        float sk = mod_lr * gk * ik;
        float2 ep = { p2.x + sp*tp.x, p2.y + sp*tp.y };
        float2 ek = { k2.x + sk*tk.x, k2.y + sk*tk.y };
        reinterpret_cast<float2*>(eff_prim + base)[lane] = ep;
        reinterpret_cast<float2*>(eff_key  + base)[lane] = ek;
        if (lane == 0) eff_dec[(size_t)b*NN + n] = sigmoidf_(dl + dm);
    }
}

// ---------------------------------------------------------------------------
// Init: copy h and prev_messages into workspace state buffers.
// ---------------------------------------------------------------------------
__global__ void init_kernel(const float* __restrict__ h_in,
                            const float* __restrict__ pm_in,
                            float* __restrict__ h0, float* __restrict__ pm0)
{
    const int total = BS*NN*DD;
    for (int e = blockIdx.x*blockDim.x + threadIdx.x; e < total;
         e += gridDim.x*blockDim.x) {
        h0[e]  = h_in[e];
        pm0[e] = pm_in[e];
    }
}

// ---------------------------------------------------------------------------
// Phase B: one wave per (b,n). Lane owns dims d=2*lane..2*lane+1.
// Pass 1: K=32 gated similarities; wave32 xor-reduction leaves sim_k on
// lane k. Pass 2: dendritic tree with shuffled broadcast of idx/weight.
// pm is double-buffered (synchronous update); h updated in place.
// ---------------------------------------------------------------------------
#define SW 8   // waves per block

__global__ __launch_bounds__(SW*32) void step_kernel(
    int t, const int* __restrict__ stridep,
    const float* __restrict__ cc,  const int* __restrict__ conn,
    const float* __restrict__ dbw, const float* __restrict__ dgw,
    const float* __restrict__ eff_prim, const float* __restrict__ eff_key,
    const float* __restrict__ eff_dec,
    float* __restrict__ hbuf,
    const float* __restrict__ pm_src, float* __restrict__ pm_dst,
    float* __restrict__ out)
{
    const int wave = threadIdx.x >> 5;
    const int lane = threadIdx.x & 31;
    const int w = blockIdx.x * SW + wave;   // 0 .. BS*NN-1
    const int n = w & (NN - 1);
    const int b = w >> 12;
    const size_t base = ((size_t)b*NN + n) * DD;

    const int stride = stridep[0];
    const bool upd = (stride > 0) && (t % stride == 0);

    float2 pmv;
    if (upd) {
        int idx_l = conn[n*KKN + lane];                 // neighbor for k = lane
        float2 ekv = reinterpret_cast<const float2*>(eff_key + base)[lane];

        // pass 1: similarities -> sigmoid weight, lane k keeps w_k
        float wk = 0.0f;
        for (int k = 0; k < KKN; ++k) {
            int idx = __shfl(idx_l, k, 32);
            float2 m = reinterpret_cast<const float2*>(
                pm_src + ((size_t)b*NN + idx)*DD)[lane];
            float p = ekv.x*m.x + ekv.y*m.y;
            #pragma unroll
            for (int off = 16; off >= 1; off >>= 1) p += __shfl_xor(p, off, 32);
            if (lane == k) wk = p;
        }
        wk = sigmoidf_(wk);

        // pass 2: dendritic tree (NB branches of BSZ, then BPG group)
        float2 gacc = {0.0f, 0.0f};
        #pragma unroll
        for (int j = 0; j < NBB; ++j) {
            float2 bracc = {0.0f, 0.0f};
            #pragma unroll
            for (int s = 0; s < BSZB; ++s) {
                int k = j*BSZB + s;
                int idx  = __shfl(idx_l, k, 32);
                float ww = __shfl(wk,   k, 32);
                float2 m = reinterpret_cast<const float2*>(
                    pm_src + ((size_t)b*NN + idx)*DD)[lane];
                float2 bw = reinterpret_cast<const float2*>(
                    dbw + (((size_t)n*NBB + j)*BSZB + s)*DD)[lane];
                bracc.x += ww * m.x * bw.x;
                bracc.y += ww * m.y * bw.y;
            }
            float2 gw = reinterpret_cast<const float2*>(
                dgw + ((size_t)n*BPGG + j)*DD)[lane];
            gacc.x += tanhf(bracc.x) * gw.x;
            gacc.y += tanhf(bracc.y) * gw.y;
        }
        float2 rec = { tanhf(gacc.x), tanhf(gacc.y) };   // mean over NG=1

        if (n < CCH) {
            float2 ccv = reinterpret_cast<const float2*>(
                cc + (((size_t)b*TT + t)*CCH + n)*DD)[lane];
            rec.x += ccv.x; rec.y += ccv.y;
        }
        float dec = eff_dec[(size_t)b*NN + n];
        float2 hv = reinterpret_cast<const float2*>(hbuf + base)[lane];
        float2 h2 = { dec*hv.x + (1.0f-dec)*rec.x,
                      dec*hv.y + (1.0f-dec)*rec.y };
        reinterpret_cast<float2*>(hbuf + base)[lane] = h2;   // in-place h
        float2 ep = reinterpret_cast<const float2*>(eff_prim + base)[lane];
        pmv.x = tanhf(h2.x * ep.x);
        pmv.y = tanhf(h2.y * ep.y);
    } else {
        pmv = reinterpret_cast<const float2*>(pm_src + base)[lane];
    }
    reinterpret_cast<float2*>(pm_dst + base)[lane] = pmv;
    if (n < CCH)
        reinterpret_cast<float2*>(out + (((size_t)b*TT + t)*CCH + n)*DD)[lane] = pmv;
}

// ---------------------------------------------------------------------------
extern "C" void kernel_launch(void* const* d_in, const int* in_sizes, int n_in,
                              void* d_out, int out_size, void* d_ws, size_t ws_size,
                              hipStream_t stream)
{
    const float* cc     = (const float*)d_in[0];
    const float* h_in   = (const float*)d_in[1];
    const float* pm_in  = (const float*)d_in[2];
    const float* tprim  = (const float*)d_in[3];
    const float* tkey   = (const float*)d_in[4];
    const float* prim   = (const float*)d_in[5];
    const float* keyp   = (const float*)d_in[6];
    const float* dlog   = (const float*)d_in[7];
    const float* dbw    = (const float*)d_in[8];
    const float* dgw    = (const float*)d_in[9];
    const float* fc1w   = (const float*)d_in[10];
    const float* fc1b   = (const float*)d_in[11];
    const float* fc2w   = (const float*)d_in[12];
    const float* fc2b   = (const float*)d_in[13];
    const float* mll    = (const float*)d_in[14];
    const int*   conn   = (const int*)d_in[15];
    const int*   stridep= (const int*)d_in[16];
    float* out = (float*)d_out;

    const size_t SZ = (size_t)BS*NN*DD;   // 2,097,152 floats
    float* eff_prim = (float*)d_ws;
    float* eff_key  = eff_prim + SZ;
    float* eff_dec  = eff_key  + SZ;
    float* hbuf     = eff_dec  + (size_t)BS*NN;
    float* pmbuf0   = hbuf     + SZ;
    float* pmbuf1   = pmbuf0   + SZ;

    phaseA_kernel<<<NN/AW, AW*32, 0, stream>>>(
        h_in, tprim, tkey, prim, keyp, dlog, fc1w, fc1b, fc2w, fc2b, mll,
        eff_prim, eff_key, eff_dec);

    init_kernel<<<4096, 256, 0, stream>>>(h_in, pm_in, hbuf, pmbuf0);

    for (int t = 0; t < TT; ++t) {
        const float* ps = (t & 1) ? pmbuf1 : pmbuf0;
        float*       pd = (t & 1) ? pmbuf0 : pmbuf1;
        step_kernel<<<(BS*NN)/SW, SW*32, 0, stream>>>(
            t, stridep, cc, conn, dbw, dgw,
            eff_prim, eff_key, eff_dec, hbuf, ps, pd, out);
    }
}
